// GSCL_motiv_14748917694892
// MI455X (gfx1250) — matrix operand order repaired and checked by reading, not compiled, not run on hardware
//
#include <hip/hip_runtime.h>

#define N_NODES 12000
#define N_PAD   12032          // 94 * 128
#define N_EDGES 384000
#define DIMF    256
// exp(x/T) = exp(2x) = exp2(x * 2*log2(e))
#define EXP2_SCALE 2.8853900817779268f
#define LN2 0.69314718055994531f

typedef __bf16 v16bf __attribute__((ext_vector_type(16)));
typedef float  v8f   __attribute__((ext_vector_type(8)));

__device__ __forceinline__ unsigned short f2bf(float f) {
    unsigned int u = __float_as_uint(f);
    unsigned int r = u + 0x7FFFu + ((u >> 16) & 1u);   // round-to-nearest-even
    return (unsigned short)(r >> 16);
}

// ---------------------------------------------------------------------------
// Generic dense GEMM: C[M,Nc] = act(A[M,K] @ B[K,Nc] + bias)
// 256 threads, 64-col slab of B in LDS (K chunked by 256 -> 64KB), 64 rows/block
// act: 0=none 1=relu 2=elu
// ---------------------------------------------------------------------------
#define GK_TN 64
#define GK_KC 256
__global__ void gemm_bias_act(const float* __restrict__ A, const float* __restrict__ B,
                              const float* __restrict__ bias, float* __restrict__ C,
                              int M, int K, int Nc, int act) {
    __shared__ float Bs[GK_KC * GK_TN];
    const int c  = threadIdx.x & 63;
    const int rs = threadIdx.x >> 6;            // 0..3
    const int colBase = blockIdx.y * GK_TN;
    const int col = colBase + c;
    const int rowStart = blockIdx.x * 64;

    float sum[16];
#pragma unroll
    for (int i = 0; i < 16; ++i) sum[i] = 0.f;

    for (int kc = 0; kc < K; kc += GK_KC) {
        const int kLen = min(GK_KC, K - kc);
        __syncthreads();
        for (int idx = threadIdx.x; idx < kLen * GK_TN; idx += 256) {
            const int k = idx >> 6, cc = idx & 63;
            const int gcol = colBase + cc;
            Bs[idx] = (gcol < Nc) ? B[(size_t)(kc + k) * Nc + gcol] : 0.f;
        }
        __syncthreads();
#pragma unroll 1
        for (int rg = 0; rg < 16; ++rg) {
            const int row = rowStart + rg * 4 + rs;
            if (row < M) {
                const float* a = A + (size_t)row * K + kc;
                float s = 0.f;
                for (int k = 0; k < kLen; ++k) s = fmaf(a[k], Bs[k * GK_TN + c], s);
                sum[rg] += s;
            }
        }
    }
    if (col < Nc) {
#pragma unroll 1
        for (int rg = 0; rg < 16; ++rg) {
            const int row = rowStart + rg * 4 + rs;
            if (row < M) {
                float s = sum[rg] + (bias ? bias[col] : 0.f);
                if (act == 1)      s = fmaxf(s, 0.f);
                else if (act == 2) s = (s > 0.f) ? s : (__expf(s) - 1.f);
                C[(size_t)row * Nc + col] = s;
            }
        }
    }
}

// ---------------------------------------------------------------------------
// SpMM: Y[rows[e],:] += vals[e] * X[cols[e],:]   (D = 256), one wave per edge
// ---------------------------------------------------------------------------
__global__ void spmm_edges(const int* __restrict__ rows, const int* __restrict__ cols,
                           const float* __restrict__ vals, const float* __restrict__ X,
                           float* __restrict__ Y, int nE) {
    const int e = blockIdx.x * 8 + (threadIdx.x >> 5);
    const int lane = threadIdx.x & 31;
    if (e >= nE) return;
    const int r = rows[e], cdx = cols[e];
    const float v = vals[e];
    const float* xp = X + (size_t)cdx * DIMF;
    float* yp = Y + (size_t)r * DIMF;
#pragma unroll
    for (int i = 0; i < 8; ++i) {
        const int k = lane + 32 * i;
        atomicAdd(&yp[k], v * xp[k]);
    }
}

// ---------------------------------------------------------------------------
// Elementwise Y = act(Y + bias[col]) over M x 256
// ---------------------------------------------------------------------------
__global__ void bias_act_ew(float* __restrict__ Y, const float* __restrict__ bias,
                            int total, int act) {
    const int i = blockIdx.x * 256 + threadIdx.x;
    if (i < total) {
        float v = Y[i] + bias[i & (DIMF - 1)];
        if (act == 1) v = fmaxf(v, 0.f);
        Y[i] = v;
    }
}

// ---------------------------------------------------------------------------
// Row-normalize Z -> bf16 ZN, and d[i] = exp(dot(zn_i,zn_i)/T). One wave/row.
// ---------------------------------------------------------------------------
__global__ void normalize_rows(const float* __restrict__ Z, unsigned short* __restrict__ ZN,
                               float* __restrict__ Dd) {
    const int lane = threadIdx.x & 31;
    const int row = blockIdx.x * 8 + (threadIdx.x >> 5);
    if (row >= N_NODES) return;
    const float* z = Z + (size_t)row * DIMF;
    float vals[8], ss = 0.f;
#pragma unroll
    for (int i = 0; i < 8; ++i) { vals[i] = z[lane + 32 * i]; ss = fmaf(vals[i], vals[i], ss); }
#pragma unroll
    for (int m = 16; m >= 1; m >>= 1) ss += __shfl_xor(ss, m, 32);
    const float inv = 1.f / fmaxf(sqrtf(ss), 1e-12f);
    unsigned short* o = ZN + (size_t)row * DIMF;
#pragma unroll
    for (int i = 0; i < 8; ++i) o[lane + 32 * i] = f2bf(vals[i] * inv);
    if (lane == 0) {
        const float dot = ss * inv * inv;          // ~1.0
        Dd[row] = __builtin_amdgcn_exp2f(dot * EXP2_SCALE);
    }
}

// ---------------------------------------------------------------------------
// Gram row-sum: RS[i] += sum_j exp( dot(zn_i, zn_j) / T ) over a j-chunk.
//
// 8 waves/block; wave w owns rows [blockIdx.x*128 + w*16), all K=256 of the
// A operand preloaded in 8 x v16bf. 16-col B tiles (B = ZN^T => a B tile is
// 16 zn rows = 8KB) staged cooperatively into LDS, double buffered, with
// prefetch distance 2 held in PARITY-INDEXED registers (pf[t&1]) + unroll-2,
// so each half uses a fixed register set (no rotation copies / waits).
//
// All 8 B chunks are ds_load'ed into independent registers before the 8-WMMA
// chain -> DS queue stays full, dscnt waits staged; the previous tile's
// exp/accumulate (TRANS + VALU) co-executes under the current WMMA chain.
//
// __launch_bounds__(256, 1): the loop is software-pipelined and only ever
// runs 2 waves/SIMD per block anyway -- raise the per-wave VGPR budget so
// the whole a[]/ub[]/pf[] working set stays resident (round-4 codegen
// spilled a[0] to scratch and reloaded it every iteration).
//
// LDS layout is fragment-order: lo[k][lane] / hi[k][lane] 16B chunks ->
// ds_load_b128 lane stride = 4 DWORDs; 32 lanes x 4 DWORDs tile the 64 banks
// exactly twice (optimal).
//
// Row sums accumulate PER LANE; one 16-lane butterfly in the epilogue.
// ---------------------------------------------------------------------------
#define JTILES_PER_BLOCK 30    // 750 real j-tiles = 25 * 30
__global__ void __launch_bounds__(256, 1)
gram_rowsum(const unsigned short* __restrict__ ZN, float* __restrict__ RS) {
    // [buf][ lo:0 / hi:2048 + (k*32+lane)*8 ushorts ]
    __shared__ unsigned short Bt[2][4096];

    const int tid  = threadIdx.x;
    const int wave = tid >> 5;
    const int lane = tid & 31;
    const int hf   = lane >> 4;        // half-wave select
    const int l15  = lane & 15;
    const int rowBase = blockIdx.x * 128 + wave * 16;
    const int myRow   = rowBase + l15;            // < N_PAD always

    // A-matrix 16x32 bf16 layout: lanes 0-15 / 16-31 share M=l15;
    // v0-3: K = hf*8 + 0..7 ; v4-7: K = 16 + hf*8 + 0..7   (contiguous 16B each)
    const unsigned short* arow = ZN + (size_t)myRow * DIMF;
    v16bf a[8];
#pragma unroll
    for (int k = 0; k < 8; ++k) {
        union { uint4 u[2]; v16bf v; } ua;
        ua.u[0] = *(const uint4*)(arow + k * 32 + hf * 8);
        ua.u[1] = *(const uint4*)(arow + k * 32 + 16 + hf * 8);
        a[k] = ua.v;
    }

    // staging role of this thread: fragment (sk, slane)
    const int sk    = tid >> 5;          // 0..7  (k-chunk)
    const int slane = tid & 31;          // consumer lane
    const int sn    = slane & 15;
    const int shf   = slane >> 4;
    const int fragOfs = ((sk * 32 + slane) * 8); // ushort offset of 16B chunk

    const int jt0 = blockIdx.y * JTILES_PER_BLOCK;
    // per-thread global source for tile t is srcBase + t*16*DIMF
    const unsigned short* srcBase =
        ZN + (size_t)(jt0 * 16 + sn) * DIMF + shf * 8 + sk * 32;

    // --- prologue: stage tile 0 straight into buffer 0; prefetch tile 1 ---
    {
        const uint4 lo0 = *(const uint4*)srcBase;
        const uint4 hi0 = *(const uint4*)(srcBase + 16);
        *(uint4*)&Bt[0][fragOfs]        = lo0;
        *(uint4*)&Bt[0][2048 + fragOfs] = hi0;
    }
    // pf[t & 1] holds tile t's fragment while it waits to be committed to LDS
    uint4 pfLo[2], pfHi[2];
    {
        const unsigned short* src = srcBase + (size_t)16 * DIMF;   // tile 1
        pfLo[1] = *(const uint4*)src;
        pfHi[1] = *(const uint4*)(src + 16);
    }
    __syncthreads();

    float rowAcc[8];
#pragma unroll
    for (int r = 0; r < 8; ++r) rowAcc[r] = 0.f;

#pragma unroll 2
    for (int jt = 0; jt < JTILES_PER_BLOCK; ++jt) {
        const int cur = jt & 1;

        // (1) issue global loads for tile jt+2 into pf[jt&1]
        //     (slot jt&1 held tile jt, already committed last iteration)
        if ((jt + 2) < JTILES_PER_BLOCK) {
            const unsigned short* src = srcBase + (size_t)(jt + 2) * 16 * DIMF;
            pfLo[cur] = *(const uint4*)src;
            pfHi[cur] = *(const uint4*)(src + 16);
        }

        // (2) commit tile jt+1 (loaded a full iteration ago) to the other buf
        if ((jt + 1) < JTILES_PER_BLOCK) {
            *(uint4*)&Bt[cur ^ 1][fragOfs]        = pfLo[cur ^ 1];
            *(uint4*)&Bt[cur ^ 1][2048 + fragOfs] = pfHi[cur ^ 1];
        }

        // (3) load ALL 8 B chunks into independent registers (DS queue full)
        union { uint4 u[2]; v16bf v; } ub[8];
#pragma unroll
        for (int k = 0; k < 8; ++k) {
            const int ofs = (k * 32 + lane) * 8;
            ub[k].u[0] = *(const uint4*)&Bt[cur][ofs];
            ub[k].u[1] = *(const uint4*)&Bt[cur][2048 + ofs];
        }

        // (4) 8 back-to-back WMMAs
        v8f acc = {};
#pragma unroll
        for (int k = 0; k < 8; ++k)
            acc = __builtin_amdgcn_wmma_f32_16x16x32_bf16(
                false, a[k], false, ub[k].v, (short)0, acc, false, false);

        // (5) C layout: vgpr r of lane l -> row = r + 8*hf, col = l15.
        //     Per-lane accumulation only; cross-lane reduce in epilogue.
#pragma unroll
        for (int r = 0; r < 8; ++r)
            rowAcc[r] += __builtin_amdgcn_exp2f(acc[r] * EXP2_SCALE);

        // (6) block-sync: stores of tile jt+1 visible to all waves
        __syncthreads();
    }

    // epilogue: sum each row's 16 column-lanes (xor < 16 stays in the half)
#pragma unroll
    for (int r = 0; r < 8; ++r) {
        float v = rowAcc[r];
        v += __shfl_xor(v, 1, 32);
        v += __shfl_xor(v, 2, 32);
        v += __shfl_xor(v, 4, 32);
        v += __shfl_xor(v, 8, 32);
        if (l15 == 0)
            atomicAdd(&RS[rowBase + hf * 8 + r], v);
    }
}

// ---------------------------------------------------------------------------
// Loss: loss_i = log(RS_i + d_i) - log(d_i); two-stage mean reduction.
// ---------------------------------------------------------------------------
__global__ void loss_partial(const float* __restrict__ RS, const float* __restrict__ Dd,
                             float* __restrict__ part, int M) {
    __shared__ float s[256];
    const int i = blockIdx.x * 256 + threadIdx.x;
    float v = 0.f;
    if (i < M) {
        const float d = Dd[i];
        v = (__builtin_amdgcn_logf(RS[i] + d) - __builtin_amdgcn_logf(d)) * LN2;
    }
    s[threadIdx.x] = v; __syncthreads();
    for (int o = 128; o > 0; o >>= 1) {
        if (threadIdx.x < o) s[threadIdx.x] += s[threadIdx.x + o];
        __syncthreads();
    }
    if (threadIdx.x == 0) part[blockIdx.x] = s[0];
}

__global__ void loss_final(const float* __restrict__ part, int nPart,
                           float* __restrict__ out, float invN) {
    __shared__ float s[256];
    float v = 0.f;
    for (int i = threadIdx.x; i < nPart; i += 256) v += part[i];
    s[threadIdx.x] = v; __syncthreads();
    for (int o = 128; o > 0; o >>= 1) {
        if (threadIdx.x < o) s[threadIdx.x] += s[threadIdx.x + o];
        __syncthreads();
    }
    if (threadIdx.x == 0) out[0] = s[0] * invN;
}

// ---------------------------------------------------------------------------
extern "C" void kernel_launch(void* const* d_in, const int* in_sizes, int n_in,
                              void* d_out, int out_size, void* d_ws, size_t ws_size,
                              hipStream_t stream) {
    const float* feat  = (const float*)d_in[0];
    const int*   erow  = (const int*)  d_in[1];
    const int*   ecol  = (const int*)  d_in[2];
    const float* evals = (const float*)d_in[3];
    const float* w1  = (const float*)d_in[4];  const float* b1  = (const float*)d_in[5];
    const float* w2  = (const float*)d_in[6];  const float* b2  = (const float*)d_in[7];
    const float* gw1 = (const float*)d_in[8];  const float* gb1 = (const float*)d_in[9];
    const float* gw2 = (const float*)d_in[10]; const float* gb2 = (const float*)d_in[11];
    const float* pw1 = (const float*)d_in[12]; const float* pb1 = (const float*)d_in[13];
    const float* pw2 = (const float*)d_in[14]; const float* pb2 = (const float*)d_in[15];

    char* ws = (char*)d_ws;
    size_t off = 0;
    auto carve = [&](size_t bytes) -> void* {
        void* p = ws + off;
        off = (off + bytes + 255) & ~(size_t)255;
        return p;
    };
    const size_t big = (size_t)N_NODES * DIMF * sizeof(float);      // 12.3 MB
    float* B0 = (float*)carve(big);                                 // XW -> HW -> Z
    float* B1 = (float*)carve(big);                                 // agg1/h -> agg2/logits
    float* H1 = (float*)carve((size_t)N_NODES * 64 * sizeof(float));
    float* X  = (float*)carve((size_t)N_NODES * 32 * sizeof(float));
    float* P1 = (float*)carve((size_t)N_NODES * 128 * sizeof(float));
    unsigned short* ZN = (unsigned short*)carve((size_t)N_PAD * DIMF * sizeof(unsigned short));
    float* Dd   = (float*)carve((size_t)N_NODES * sizeof(float));
    float* RS   = (float*)carve((size_t)N_PAD * sizeof(float));
    float* PART = (float*)carve(64 * sizeof(float));

    const dim3 blk(256);

    // 1) H1 = relu(feat @ w1 + b1)        [12000,512]x[512,64]
    gemm_bias_act<<<dim3(188, 1), blk, 0, stream>>>(feat, w1, b1, H1, N_NODES, 512, 64, 1);
    // 2) X = H1 @ w2 + b2                 [12000,64]x[64,32]
    gemm_bias_act<<<dim3(188, 1), blk, 0, stream>>>(H1, w2, b2, X, N_NODES, 64, 32, 0);
    // 3) XW = X @ gw1                     [12000,32]x[32,256]
    gemm_bias_act<<<dim3(188, 4), blk, 0, stream>>>(X, gw1, nullptr, B0, N_NODES, 32, 256, 0);
    // 4) agg1 = spmm(A, XW)
    hipMemsetAsync(B1, 0, big, stream);
    spmm_edges<<<dim3(N_EDGES / 8), blk, 0, stream>>>(erow, ecol, evals, B0, B1, N_EDGES);
    // 5) h = relu(agg1 + gb1)   (in place)
    bias_act_ew<<<dim3(N_NODES), blk, 0, stream>>>(B1, gb1, N_NODES * DIMF, 1);
    // 6) HW = h @ gw2                     [12000,256]x[256,256]
    gemm_bias_act<<<dim3(188, 4), blk, 0, stream>>>(B1, gw2, nullptr, B0, N_NODES, 256, 256, 0);
    // 7) agg2 = spmm(A, HW)
    hipMemsetAsync(B1, 0, big, stream);
    spmm_edges<<<dim3(N_EDGES / 8), blk, 0, stream>>>(erow, ecol, evals, B0, B1, N_EDGES);
    // 8) logits = agg2 + gb2    (in place)
    bias_act_ew<<<dim3(N_NODES), blk, 0, stream>>>(B1, gb2, N_NODES * DIMF, 0);
    // 9) P1 = elu(logits @ pw1 + pb1)     [12000,256]x[256,128]
    gemm_bias_act<<<dim3(188, 2), blk, 0, stream>>>(B1, pw1, pb1, P1, N_NODES, 256, 128, 2);
    // 10) Z = P1 @ pw2 + pb2              [12000,128]x[128,256]
    gemm_bias_act<<<dim3(188, 4), blk, 0, stream>>>(P1, pw2, pb2, B0, N_NODES, 128, 256, 0);
    // 11) zn (bf16) + d; zero pad rows of ZN and RS
    hipMemsetAsync(ZN + (size_t)N_NODES * DIMF, 0,
                   (size_t)(N_PAD - N_NODES) * DIMF * sizeof(unsigned short), stream);
    normalize_rows<<<dim3(N_NODES / 8), blk, 0, stream>>>(B0, ZN, Dd);
    hipMemsetAsync(RS, 0, (size_t)N_PAD * sizeof(float), stream);
    // 12) gram row-sums via bf16 WMMA: grid (94 row-stripes) x (25 j-chunks)
    gram_rowsum<<<dim3(N_PAD / 128, 750 / JTILES_PER_BLOCK), blk, 0, stream>>>(ZN, RS);
    // 13) loss mean
    const int nPart = (N_NODES + 255) / 256;   // 47
    loss_partial<<<dim3(nPart), blk, 0, stream>>>(RS, Dd, PART, N_NODES);
    loss_final<<<dim3(1), blk, 0, stream>>>(PART, nPart, (float*)d_out, 1.f / N_NODES);
}
